// BlockStackingSGN_62225486185130
// MI455X (gfx1250) — compile-verified, matrix operand order checked by reading, not run on hardware
//
#include <hip/hip_runtime.h>
#include <cstdint>

#define NB      8
#define HID     256
#define BATCH   4096
#define IND     24
#define MTILE   128
#define LDR     264            // LDS row stride in bf16 elems (pad 8 -> conflict-friendly)
#define TILE_ELEMS (MTILE * LDR)
#define THREADS 256

typedef __attribute__((ext_vector_type(16))) __bf16 v16bf;
typedef __attribute__((ext_vector_type(8)))  float  v8f;

union Frag { v16bf v; uint4 q[2]; };
union U4   { uint4 q; __bf16 h[8]; };

// ---------------------------------------------------------------------------
// Wave-level 128x256 GEMM: D(MTILE x 256) = act(A, MTILE x K) * Wpacked(K x 256)
// A: LDS bf16 tile, row stride LDR.  W: packed global bf16 fragments.
// Packed layout: block (nt,kt) = 512 bf16; lane l holds n = nt*16+(l&15),
// elements e: k_local = e + (e&8) + ((l&16)>>1)  (matches ISA 16-bit A/B striping).
// K and HASBIAS are compile-time: the kt loop fully unrolls so every fragment
// gets distinct VGPRs -> scheduler can overlap global loads with WMMAs.
// ---------------------------------------------------------------------------
template<bool RELU, bool HASBIAS, int K>
__device__ __forceinline__ void gemm_tile(const __bf16* __restrict__ A,
                                          const __bf16* __restrict__ W,
                                          const float*  __restrict__ bias,
                                          __bf16* __restrict__ D) {
  constexpr int Kt = K >> 5;
  constexpr int bs = Kt * 512;          // element stride between n-tile blocks
  const int tid   = threadIdx.x;
  const int lane  = tid & 31;
  const int wave  = tid >> 5;
  const int m0    = wave * 16;
  const int arow  = m0 + (lane & 15);
  const int khalf = (lane & 16) >> 1;   // lanes 16..31 hold K+8 / K+24
  const int mo    = m0 + ((lane & 16) >> 1);  // D row base (M or M+8)

  for (int nt0 = 0; nt0 < 16; nt0 += 4) {
    v8f acc[4] = {};
#pragma unroll
    for (int kt = 0; kt < Kt; ++kt) {
      Frag a;
      const __bf16* ap = A + arow * LDR + kt * 32 + khalf;
      a.q[0] = *(const uint4*)(ap);        // K+0..7   (or +8..15)
      a.q[1] = *(const uint4*)(ap + 16);   // K+16..23 (or +24..31)
      const __bf16* wp = W + (nt0 * Kt + kt) * 512 + lane * 16;
#pragma unroll
      for (int t = 0; t < 4; ++t) {
        Frag b;
        b.q[0] = *(const uint4*)(wp + t * bs);
        b.q[1] = *(const uint4*)(wp + t * bs + 8);
        acc[t] = __builtin_amdgcn_wmma_f32_16x16x32_bf16(
            false, a.v, false, b.v, (short)0, acc[t], false, false);
      }
    }
#pragma unroll
    for (int t = 0; t < 4; ++t) {
      const int n = (nt0 + t) * 16 + (lane & 15);
      float bn = 0.0f;
      if (HASBIAS) bn = bias[n];
#pragma unroll
      for (int r = 0; r < 8; ++r) {
        float v = acc[t][r] + bn;
        if (RELU) v = v > 0.0f ? v : 0.0f;
        D[(mo + r) * LDR + n] = (__bf16)v;
      }
    }
  }
}

// Vectorized 256-wide dot + sigmoid: uint4 LDS reads (8 bf16) x float4 weights.
__device__ __forceinline__ float head_sigmoid(const __bf16* __restrict__ row,
                                              const float* __restrict__ w, float b) {
  float z = b;
#pragma unroll
  for (int c = 0; c < HID / 8; ++c) {
    U4 q; q.q = *(const uint4*)(row + c * 8);
    const float4 w0 = *(const float4*)(w + c * 8);
    const float4 w1 = *(const float4*)(w + c * 8 + 4);
    z += (float)q.h[0] * w0.x + (float)q.h[1] * w0.y +
         (float)q.h[2] * w0.z + (float)q.h[3] * w0.w +
         (float)q.h[4] * w1.x + (float)q.h[5] * w1.y +
         (float)q.h[6] * w1.z + (float)q.h[7] * w1.w;
  }
  return 1.0f / (1.0f + __expf(-z));
}

// ---------------------------------------------------------------------------
// Pack fp32 weight (Ksrc x 256 row-major, possibly stacked nmat) into bf16
// WMMA-B fragment layout, zero-padding K to Kpad.
// ---------------------------------------------------------------------------
__global__ void pack_weight_kernel(const float* __restrict__ W, __bf16* __restrict__ dst,
                                   int Ksrc, int Kpad, long src_stride, long dst_stride) {
  const int mat = blockIdx.y;
  const int idx = blockIdx.x * blockDim.x + threadIdx.x;
  if (idx >= Kpad * 256) return;
  const int Kt     = Kpad >> 5;
  const int block  = idx >> 9;
  const int within = idx & 511;
  const int lane   = within >> 4;
  const int e      = within & 15;
  const int nt     = block / Kt;
  const int kt     = block - nt * Kt;
  const int n      = nt * 16 + (lane & 15);
  const int klocal = e + (e & 8) + ((lane & 16) >> 1);
  const int k      = kt * 32 + klocal;
  const float v = (k < Ksrc) ? W[mat * src_stride + (long)k * 256 + n] : 0.0f;
  dst[mat * dst_stride + idx] = (__bf16)v;
}

// ---------------------------------------------------------------------------
// Kernel 1: per-block object encoder (3 GEMM layers) -> enc bf16 [NB][BATCH][HID]
// ---------------------------------------------------------------------------
__global__ void obj_enc_kernel(const float* __restrict__ x,
                               const __bf16* __restrict__ oW0p, const float* __restrict__ o_b0,
                               const __bf16* __restrict__ oW1p, const float* __restrict__ o_b1,
                               const __bf16* __restrict__ oW2p, const float* __restrict__ o_b2,
                               __bf16* __restrict__ enc) {
  extern __shared__ __bf16 smem[];
  __bf16* buf0 = smem;
  __bf16* buf1 = smem + TILE_ELEMS;
  __bf16* buf2 = smem + 2 * TILE_ELEMS;
  const int  n  = blockIdx.y;
  const long b0 = (long)blockIdx.x * MTILE;
  const int  tid = threadIdx.x;

  for (int idx = tid; idx < MTILE * 32; idx += THREADS) {   // stage x, pad K 24->32
    const int m = idx >> 5, k = idx & 31;
    const float v = (k < IND) ? x[(b0 + m) * IND + k] : 0.0f;
    buf0[m * LDR + k] = (__bf16)v;
  }
  __syncthreads();
  gemm_tile<true,  true, 32 >(buf0, oW0p + (long)n * (32 * 256),  o_b0 + n * HID, buf1);
  __syncthreads();
  gemm_tile<true,  true, 256>(buf1, oW1p + (long)n * (256 * 256), o_b1 + n * HID, buf2);
  __syncthreads();
  gemm_tile<false, true, 256>(buf2, oW2p + (long)n * (256 * 256), o_b2 + n * HID, buf0);
  __syncthreads();
  for (int idx = tid; idx < MTILE * 32; idx += THREADS) {   // coalesced b128 store
    const int m = idx >> 5, c = idx & 31;
    const uint4 q = *(const uint4*)(buf0 + m * LDR + c * 8);
    *(uint4*)(enc + (((long)n * BATCH + b0 + m) * HID + c * 8)) = q;
  }
}

// ---------------------------------------------------------------------------
// Kernel 2: clear/table heads -> d_out, and a_l/a_r projections -> ws (bf16)
// ---------------------------------------------------------------------------
__global__ void heads_kernel(const __bf16* __restrict__ enc,
                             const __bf16* __restrict__ cW0p, const float* __restrict__ c_b0,
                             const __bf16* __restrict__ cW1p, const float* __restrict__ c_b1,
                             const float* __restrict__ c_W2,  const float* __restrict__ c_b2,
                             const __bf16* __restrict__ tW0p, const float* __restrict__ t_b0,
                             const __bf16* __restrict__ tW1p, const float* __restrict__ t_b1,
                             const float* __restrict__ t_W2,  const float* __restrict__ t_b2,
                             const __bf16* __restrict__ aW0Lp, const __bf16* __restrict__ aW0Rp,
                             __bf16* __restrict__ al, __bf16* __restrict__ ar,
                             float* __restrict__ out) {
  extern __shared__ __bf16 smem[];
  __bf16* buf0 = smem;
  __bf16* buf1 = smem + TILE_ELEMS;
  __bf16* buf2 = smem + 2 * TILE_ELEMS;
  const int  n  = blockIdx.y;
  const long b0 = (long)blockIdx.x * MTILE;
  const int  tid = threadIdx.x;

  for (int idx = tid; idx < MTILE * 32; idx += THREADS) {   // load enc tile
    const int m = idx >> 5, c = idx & 31;
    *(uint4*)(buf0 + m * LDR + c * 8) =
        *(const uint4*)(enc + (((long)n * BATCH + b0 + m) * HID + c * 8));
  }
  __syncthreads();
  // clear predicate
  gemm_tile<true, true, 256>(buf0, cW0p, c_b0, buf1); __syncthreads();
  gemm_tile<true, true, 256>(buf1, cW1p, c_b1, buf2); __syncthreads();
  if (tid < MTILE)
    out[(b0 + tid) * 80 + n * 10 + 8] = head_sigmoid(buf2 + tid * LDR, c_W2, c_b2[0]);
  __syncthreads();
  // ontable predicate
  gemm_tile<true, true, 256>(buf0, tW0p, t_b0, buf1); __syncthreads();
  gemm_tile<true, true, 256>(buf1, tW1p, t_b1, buf2); __syncthreads();
  if (tid < MTILE)
    out[(b0 + tid) * 80 + n * 10 + 9] = head_sigmoid(buf2 + tid * LDR, t_W2, t_b2[0]);
  __syncthreads();
  // a_l / a_r projections (no bias / activation here)
  gemm_tile<false, false, 256>(buf0, aW0Lp, nullptr, buf1);
  gemm_tile<false, false, 256>(buf0, aW0Rp, nullptr, buf2);
  __syncthreads();
  for (int idx = tid; idx < MTILE * 32; idx += THREADS) {
    const int m = idx >> 5, c = idx & 31;
    const long off = ((long)n * BATCH + b0 + m) * HID + c * 8;
    *(uint4*)(al + off) = *(const uint4*)(buf1 + m * LDR + c * 8);
    *(uint4*)(ar + off) = *(const uint4*)(buf2 + m * LDR + c * 8);
  }
}

// ---------------------------------------------------------------------------
// Kernel 3: AonB over all 64 (i,j) pairs — the dominant GEMM
// ---------------------------------------------------------------------------
__global__ void aonb_kernel(const __bf16* __restrict__ al, const __bf16* __restrict__ ar,
                            const float* __restrict__ a_b0,
                            const __bf16* __restrict__ aW1p, const float* __restrict__ a_b1,
                            const float* __restrict__ a_W2,  const float* __restrict__ a_b2,
                            float* __restrict__ out) {
  extern __shared__ __bf16 smem[];
  __bf16* buf0 = smem;
  __bf16* buf1 = smem + TILE_ELEMS;
  const int  p  = blockIdx.y;
  const int  i  = p >> 3, j = p & 7;
  const long b0 = (long)blockIdx.x * MTILE;
  const int  tid = threadIdx.x;

  for (int idx = tid; idx < MTILE * 32; idx += THREADS) {   // ph0 = relu(al+ar+b0)
    const int m = idx >> 5, c = idx & 31;
    const int h0 = c * 8;
    U4 ql, qr, o;
    ql.q = *(const uint4*)(al + (((long)i * BATCH + b0 + m) * HID + h0));
    qr.q = *(const uint4*)(ar + (((long)j * BATCH + b0 + m) * HID + h0));
#pragma unroll
    for (int e = 0; e < 8; ++e) {
      float v = (float)ql.h[e] + (float)qr.h[e] + a_b0[h0 + e];
      o.h[e] = (__bf16)(v > 0.0f ? v : 0.0f);
    }
    *(uint4*)(buf0 + m * LDR + h0) = o.q;
  }
  __syncthreads();
  gemm_tile<true, true, 256>(buf0, aW1p, a_b1, buf1);
  __syncthreads();
  if (tid < MTILE)
    out[(b0 + tid) * 80 + i * 10 + j] = head_sigmoid(buf1 + tid * LDR, a_W2, a_b2[0]);
}

// ---------------------------------------------------------------------------
// Workspace layout (bytes)
// ---------------------------------------------------------------------------
#define WS_OW0   0L
#define WS_OW1   131072L
#define WS_OW2   1179648L
#define WS_CW0   2228224L
#define WS_CW1   2359296L
#define WS_TW0   2490368L
#define WS_TW1   2621440L
#define WS_AW1   2752512L
#define WS_AW0L  2883584L
#define WS_AW0R  3014656L
#define WS_ENC   3145728L
#define WS_AL    19922944L
#define WS_AR    36700160L

extern "C" void kernel_launch(void* const* d_in, const int* in_sizes, int n_in,
                              void* d_out, int out_size, void* d_ws, size_t ws_size,
                              hipStream_t stream) {
  (void)in_sizes; (void)n_in; (void)out_size; (void)ws_size;
  const float* x    = (const float*)d_in[0];
  const float* o_W0 = (const float*)d_in[1];
  const float* o_b0 = (const float*)d_in[2];
  const float* o_W1 = (const float*)d_in[3];
  const float* o_b1 = (const float*)d_in[4];
  const float* o_W2 = (const float*)d_in[5];
  const float* o_b2 = (const float*)d_in[6];
  const float* c_W0 = (const float*)d_in[7];
  const float* c_b0 = (const float*)d_in[8];
  const float* c_W1 = (const float*)d_in[9];
  const float* c_b1 = (const float*)d_in[10];
  const float* c_W2 = (const float*)d_in[11];
  const float* c_b2 = (const float*)d_in[12];
  const float* t_W0 = (const float*)d_in[13];
  const float* t_b0 = (const float*)d_in[14];
  const float* t_W1 = (const float*)d_in[15];
  const float* t_b1 = (const float*)d_in[16];
  const float* t_W2 = (const float*)d_in[17];
  const float* t_b2 = (const float*)d_in[18];
  const float* a_W0 = (const float*)d_in[19];
  const float* a_b0 = (const float*)d_in[20];
  const float* a_W1 = (const float*)d_in[21];
  const float* a_b1 = (const float*)d_in[22];
  const float* a_W2 = (const float*)d_in[23];
  const float* a_b2 = (const float*)d_in[24];

  char* ws = (char*)d_ws;
  __bf16* oW0p  = (__bf16*)(ws + WS_OW0);
  __bf16* oW1p  = (__bf16*)(ws + WS_OW1);
  __bf16* oW2p  = (__bf16*)(ws + WS_OW2);
  __bf16* cW0p  = (__bf16*)(ws + WS_CW0);
  __bf16* cW1p  = (__bf16*)(ws + WS_CW1);
  __bf16* tW0p  = (__bf16*)(ws + WS_TW0);
  __bf16* tW1p  = (__bf16*)(ws + WS_TW1);
  __bf16* aW1p  = (__bf16*)(ws + WS_AW1);
  __bf16* aW0Lp = (__bf16*)(ws + WS_AW0L);
  __bf16* aW0Rp = (__bf16*)(ws + WS_AW0R);
  __bf16* enc   = (__bf16*)(ws + WS_ENC);
  __bf16* al    = (__bf16*)(ws + WS_AL);
  __bf16* ar    = (__bf16*)(ws + WS_AR);

  // ---- pack all weights to bf16 fragment layout (every call: deterministic) ----
  pack_weight_kernel<<<dim3(32, 8),  256, 0, stream>>>(o_W0, oW0p, 24, 32, 24L * 256, 8192L);
  pack_weight_kernel<<<dim3(256, 8), 256, 0, stream>>>(o_W1, oW1p, 256, 256, 65536L, 65536L);
  pack_weight_kernel<<<dim3(256, 8), 256, 0, stream>>>(o_W2, oW2p, 256, 256, 65536L, 65536L);
  pack_weight_kernel<<<dim3(256, 1), 256, 0, stream>>>(c_W0, cW0p, 256, 256, 0L, 0L);
  pack_weight_kernel<<<dim3(256, 1), 256, 0, stream>>>(c_W1, cW1p, 256, 256, 0L, 0L);
  pack_weight_kernel<<<dim3(256, 1), 256, 0, stream>>>(t_W0, tW0p, 256, 256, 0L, 0L);
  pack_weight_kernel<<<dim3(256, 1), 256, 0, stream>>>(t_W1, tW1p, 256, 256, 0L, 0L);
  pack_weight_kernel<<<dim3(256, 1), 256, 0, stream>>>(a_W1, aW1p, 256, 256, 0L, 0L);
  pack_weight_kernel<<<dim3(256, 1), 256, 0, stream>>>(a_W0,             aW0Lp, 256, 256, 0L, 0L);
  pack_weight_kernel<<<dim3(256, 1), 256, 0, stream>>>(a_W0 + 256 * 256, aW0Rp, 256, 256, 0L, 0L);

  const size_t lds3 = (size_t)3 * TILE_ELEMS * sizeof(__bf16);
  const size_t lds2 = (size_t)2 * TILE_ELEMS * sizeof(__bf16);

  obj_enc_kernel<<<dim3(BATCH / MTILE, NB), THREADS, lds3, stream>>>(
      x, oW0p, o_b0, oW1p, o_b1, oW2p, o_b2, enc);

  heads_kernel<<<dim3(BATCH / MTILE, NB), THREADS, lds3, stream>>>(
      enc, cW0p, c_b0, cW1p, c_b1, c_W2, c_b2,
      tW0p, t_b0, tW1p, t_b1, t_W2, t_b2,
      aW0Lp, aW0Rp, al, ar, (float*)d_out);

  aonb_kernel<<<dim3(BATCH / MTILE, NB * NB), THREADS, lds2, stream>>>(
      al, ar, a_b0, aW1p, a_b1, a_W2, a_b2, (float*)d_out);
}
